// VARCMultiHeadAttention_56753697849555
// MI455X (gfx1250) — matrix operand order repaired and checked by reading, not compiled, not run on hardware
//
#include <hip/hip_runtime.h>

typedef __bf16 bf16;
typedef __attribute__((ext_vector_type(4)))  __bf16 bf16x4;
typedef __attribute__((ext_vector_type(8)))  __bf16 bf16x8;
typedef __attribute__((ext_vector_type(16))) __bf16 v16bf;
typedef __attribute__((ext_vector_type(8)))  float  v8f;

constexpr int kE  = 384;   // embed dim
constexpr int kS  = 2048;  // seq len
constexpr int kB  = 4;     // batch
constexpr int kH  = 8;     // heads
constexpr int kD  = 48;    // head dim (real)
constexpr int kDP = 64;    // head dim padded to 2x WMMA K (32)

// ---- CDNA5 async global->LDS path (guarded; falls back to plain copies) --

#if defined(__HIP_DEVICE_COMPILE__) && \
    __has_builtin(__builtin_amdgcn_global_load_async_to_lds_b128)
#define HAVE_ASYNC_LDS 1
#else
#define HAVE_ASYNC_LDS 0
#endif

#if HAVE_ASYNC_LDS
// builtin is typed by payload width: b128 => v4i32 pointers
typedef int v4i __attribute__((vector_size(16)));
typedef __attribute__((address_space(1))) v4i gv4i;   // global (AS1)
typedef __attribute__((address_space(3))) v4i lv4i;   // LDS (AS3)
__device__ __forceinline__ void async_cp16(const __bf16* gsrc, __bf16* ldst) {
  __builtin_amdgcn_global_load_async_to_lds_b128(
      (gv4i*)gsrc, (lv4i*)ldst, /*offset=*/0, /*cpol=*/0);
}
#endif

template <int N>
__device__ __forceinline__ void wait_asynccnt() {
#if HAVE_ASYNC_LDS
#if __has_builtin(__builtin_amdgcn_s_wait_asynccnt)
  __builtin_amdgcn_s_wait_asynccnt(N);
#else
  asm volatile("s_wait_asynccnt %0" ::"i"(N) : "memory");
#endif
#endif
}

// ---- helpers -------------------------------------------------------------

__device__ __forceinline__ bf16 f2bf(float f) {
  unsigned u = __builtin_bit_cast(unsigned, f);
  unsigned r = u + 0x7FFFu + ((u >> 16) & 1u);   // round-to-nearest-even
  unsigned short h = (unsigned short)(r >> 16);
  return __builtin_bit_cast(bf16, h);
}

union frag16 { bf16x8 h[2]; v16bf v; };

__device__ __forceinline__ v16bf ld2x8(const bf16* p0, const bf16* p1) {
  frag16 f;
  f.h[0] = *(const bf16x8*)p0;
  f.h[1] = *(const bf16x8*)p1;
  return f.v;
}

__device__ __forceinline__ v8f wmma_bf16(v16bf a, v16bf b, v8f c) {
  // D = A(16x32 bf16) x B(32x16 bf16) + C(16x16 f32)
  return __builtin_amdgcn_wmma_f32_16x16x32_bf16(
      /*neg_a=*/false, a, /*neg_b=*/false, b,
      /*c_mod=*/(short)0, c, /*reuse_a=*/false, /*reuse_b=*/false);
}

// ---- kernel 0: weights fp32 -> bf16 --------------------------------------

__global__ __launch_bounds__(256) void cvt_weights_kernel(
    const float* __restrict__ W0, const float* __restrict__ W1,
    const float* __restrict__ W2, const float* __restrict__ W3,
    bf16* __restrict__ dst) {
  const int groups = (kE * kE) / 4;            // float4 groups per matrix
  int i = blockIdx.x * 256 + threadIdx.x;      // 0 .. 4*groups-1
  int which = i / groups;
  int j = i - which * groups;
  const float* W = (which == 0) ? W0 : (which == 1) ? W1 : (which == 2) ? W2 : W3;
  float4 f = ((const float4*)W)[j];
  bf16x4 h;
  h.x = f2bf(f.x); h.y = f2bf(f.y); h.z = f2bf(f.z); h.w = f2bf(f.w);
  ((bf16x4*)dst)[(size_t)which * groups + j] = h;
}

// ---- kernel 1: fused QKV projection --------------------------------------
// X[8192,384] f32 -> Q,K [B,H,S,64] bf16 ; Vt [B,H,64,S] bf16 (transposed)

__global__ __launch_bounds__(256) void qkv_kernel(
    const float* __restrict__ X,
    const bf16* __restrict__ Wq, const bf16* __restrict__ Wk,
    const bf16* __restrict__ Wv,
    const float* __restrict__ bq, const float* __restrict__ bk,
    const float* __restrict__ bv,
    bf16* __restrict__ Q, bf16* __restrict__ K, bf16* __restrict__ Vt) {
  __shared__ __align__(16) bf16 xs[16 * kE];   // 12 KB staged A tile (bf16)

  const int tid = threadIdx.x;
  const int rt  = blockIdx.x;                  // 16-row tile of X

  // cooperative stage + convert: 6144 floats (conversion forces VGPR path)
  const float4* Xv = (const float4*)(X + (size_t)rt * 16 * kE);
  for (int i = tid; i < 16 * kE / 4; i += 256) {
    float4 f = Xv[i];
    bf16x4 h;
    h.x = f2bf(f.x); h.y = f2bf(f.y); h.z = f2bf(f.z); h.w = f2bf(f.w);
    *(bf16x4*)(xs + i * 4) = h;
  }
  __syncthreads();

  const int w    = tid >> 5;
  const int lane = tid & 31;
  const int m    = lane & 15;
  const int half = lane >> 4;

  for (int ti = 0; ti < 9; ++ti) {             // 8 waves x 9 = 72 tiles
    int t   = w * 9 + ti;
    int mat = t / 24;                          // 0=q 1=k 2=v
    int n0  = (t % 24) * 16;
    const bf16*  W    = (mat == 0) ? Wq : (mat == 1) ? Wk : Wv;
    const float* bias = (mat == 0) ? bq : (mat == 1) ? bk : bv;

    int   c  = n0 + m;                         // output column (lane-fixed)
    float bv_ = bias[c];
    v8f acc = {bv_, bv_, bv_, bv_, bv_, bv_, bv_, bv_};

#pragma unroll
    for (int kk = 0; kk < kE; kk += 32) {
      v16bf a = ld2x8(xs + m * kE + kk + half * 16,
                      xs + m * kE + kk + half * 16 + 8);
      const bf16* wr = W + (size_t)c * kE + kk + half * 16;
      v16bf bfr = ld2x8(wr, wr + 8);
      acc = wmma_bf16(a, bfr, acc);
    }

    int hh = c / kD;                           // head
    int dd = c - hh * kD;                      // dim within head
#pragma unroll
    for (int r = 0; r < 8; ++r) {
      int R  = rt * 16 + r + half * 8;         // global row in [0,8192)
      int bb = R >> 11;
      int ss = R & (kS - 1);
      bf16 val = f2bf(acc[r]);
      if (mat == 0) {
        Q[(((size_t)bb * kH + hh) * kS + ss) * kDP + dd] = val;
      } else if (mat == 1) {
        K[(((size_t)bb * kH + hh) * kS + ss) * kDP + dd] = val;
      } else {
        Vt[(((size_t)bb * kH + hh) * kDP + dd) * kS + ss] = val;
      }
    }
  }
}

// ---- kernel 2: flash attention -------------------------------------------
// Block = one (b,h), 8 waves = 8 consecutive 16-row q-tiles. K/V blocks are
// double-buffered in LDS via async global->LDS copies (ASYNCcnt pipelined).

__global__ __launch_bounds__(256) void attn_kernel(
    const bf16* __restrict__ Q, const bf16* __restrict__ K,
    const bf16* __restrict__ Vt, const unsigned char* __restrict__ mask,
    bf16* __restrict__ Attn) {
  __shared__ __align__(16) bf16 Klds[2][32 * kDP];  // [key][d]   2 x 4 KB
  __shared__ __align__(16) bf16 Vlds[2][kDP * 32];  // [d][key]   2 x 4 KB
  __shared__ __align__(16) bf16 Plds[8][16 * 32];   // per-wave P buf, 8 KB

  const int tid  = threadIdx.x;
  const int w    = tid >> 5;
  const int lane = tid & 31;
  const int m    = lane & 15;
  const int half = lane >> 4;

  const int bh = blockIdx.x >> 4;              // (b*H + h), shared by block
  const int qc = blockIdx.x & 15;              // q-chunk of 8 tiles
  const int qt = qc * 8 + w;                   // this wave's 16-row q tile
  const int b  = bh >> 3;
  const int h  = bh & 7;

  const bf16* Qb = Q  + ((size_t)bh * kS + (size_t)qt * 16) * kDP;
  const bf16* Kb = K  + (size_t)bh * kS * kDP;
  const bf16* Vb = Vt + (size_t)bh * kDP * kS;
  const unsigned char* mb = mask + (size_t)b * kS;

  // cooperative K/V block staging: 4 KB K + 4 KB V = one b128 each per thread
  auto stage_kv = [&](int key0, int buf) {
    const bf16* gk = Kb + (size_t)(key0 + (tid >> 3)) * kDP + (tid & 7) * 8;
    bf16*       lk = &Klds[buf][(tid >> 3) * kDP + (tid & 7) * 8];
    const bf16* gv = Vb + (size_t)(tid >> 2) * kS + key0 + (tid & 3) * 8;
    bf16*       lv = &Vlds[buf][(tid >> 2) * 32 + (tid & 3) * 8];
#if HAVE_ASYNC_LDS
    async_cp16(gk, lk);
    async_cp16(gv, lv);
#else
    *(uint4*)lk = *(const uint4*)gk;
    *(uint4*)lv = *(const uint4*)gv;
#endif
  };

  // Q A-fragments, reused for all key blocks (K dim = 64 = 2 x 32)
  const bf16* qr = Qb + (size_t)m * kDP;
  v16bf aQ0 = ld2x8(qr + half * 16,      qr + half * 16 + 8);
  v16bf aQ1 = ld2x8(qr + 32 + half * 16, qr + 40 + half * 16);

  float mi[8], li[8];
  v8f o0 = {0, 0, 0, 0, 0, 0, 0, 0}, o1 = o0, o2 = o0;
#pragma unroll
  for (int r = 0; r < 8; ++r) { mi[r] = -3.0e38f; li[r] = 0.0f; }

  const float scale = 0.14433756729740643f;    // 1/sqrt(48)
  bf16* pl = Plds[w];

  stage_kv(0, 0);                              // prologue: fill buffer 0

  for (int kb = 0; kb < kS / 32; ++kb) {
    const int cur  = kb & 1;
    const int key0 = kb * 32;

    if (kb + 1 < kS / 32) {
      stage_kv(key0 + 32, cur ^ 1);            // overlap next-block copy
      __builtin_prefetch(mb + key0 + 32, 0, 1);
      wait_asynccnt<2>();                      // current buffer's copies done
    } else {
      wait_asynccnt<0>();
    }
    __syncthreads();                           // buffer `cur` visible to all

    // scores: two 16x16 tiles covering keys [key0, key0+32), K from LDS
    const bf16* kr0 = &Klds[cur][m * kDP + half * 16];
    const bf16* kr1 = &Klds[cur][(16 + m) * kDP + half * 16];
    v8f s0 = {0, 0, 0, 0, 0, 0, 0, 0}, s1 = s0;
    s0 = wmma_bf16(aQ0, ld2x8(kr0,      kr0 + 8),  s0);
    s0 = wmma_bf16(aQ1, ld2x8(kr0 + 32, kr0 + 40), s0);
    s1 = wmma_bf16(aQ0, ld2x8(kr1,      kr1 + 8),  s1);
    s1 = wmma_bf16(aQ1, ld2x8(kr1 + 32, kr1 + 40), s1);

    const bool msk0 = mb[key0 + m]      != 0;  // reference: True -> -inf
    const bool msk1 = mb[key0 + 16 + m] != 0;

    float p0[8], p1[8];
#pragma unroll
    for (int r = 0; r < 8; ++r) {
      float x0 = msk0 ? -1.0e30f : s0[r] * scale;
      float x1 = msk1 ? -1.0e30f : s1[r] * scale;
      float t = fmaxf(x0, x1);                 // row max across 16 lanes
#pragma unroll
      for (int off = 1; off < 16; off <<= 1)
        t = fmaxf(t, __shfl_xor(t, off, 32));
      float mn   = fmaxf(mi[r], t);
      float corr = __expf(mi[r] - mn);
      mi[r] = mn;
      float e0 = __expf(x0 - mn);
      float e1 = __expf(x1 - mn);
      float rs = e0 + e1;                      // row sum across 16 lanes
#pragma unroll
      for (int off = 1; off < 16; off <<= 1)
        rs += __shfl_xor(rs, off, 32);
      li[r] = li[r] * corr + rs;
      o0[r] *= corr; o1[r] *= corr; o2[r] *= corr;
      p0[r] = e0; p1[r] = e1;
    }

    // P (C/D layout) -> LDS -> A layout (wave-local round trip)
#pragma unroll
    for (int r = 0; r < 8; ++r) {
      int row = r + half * 8;
      pl[row * 32 + m]      = f2bf(p0[r]);
      pl[row * 32 + 16 + m] = f2bf(p1[r]);
    }
    asm volatile("s_wait_dscnt 0x0" ::: "memory");

    const bf16* pr = pl + m * 32;
    v16bf aP = ld2x8(pr + half * 16, pr + half * 16 + 8);

    // O += P x V for d-tiles 0..2 (tile 3 is pad, skipped), V from LDS
    const bf16* vr0 = &Vlds[cur][m * 32 + half * 16];
    const bf16* vr1 = vr0 + 16 * 32;
    const bf16* vr2 = vr0 + 32 * 32;
    o0 = wmma_bf16(aP, ld2x8(vr0, vr0 + 8), o0);
    o1 = wmma_bf16(aP, ld2x8(vr1, vr1 + 8), o1);
    o2 = wmma_bf16(aP, ld2x8(vr2, vr2 + 8), o2);

    __syncthreads();                           // all waves done with `cur`
  }

  // epilogue: normalize + store bf16 attention output [B,S,E]
#pragma unroll
  for (int r = 0; r < 8; ++r) {
    float  inv = 1.0f / li[r];
    int    q   = qt * 16 + r + half * 8;
    size_t row = (size_t)b * kS + q;
    Attn[row * kE + h * kD + m]      = f2bf(o0[r] * inv);
    Attn[row * kE + h * kD + 16 + m] = f2bf(o1[r] * inv);
    Attn[row * kE + h * kD + 32 + m] = f2bf(o2[r] * inv);
  }
}

// ---- kernel 3: output projection -----------------------------------------
// Attn[8192,384] bf16 @ Wo^T + bo -> out f32; tile staged via async copies.

__global__ __launch_bounds__(256) void oproj_kernel(
    const bf16* __restrict__ A, const bf16* __restrict__ Wo,
    const float* __restrict__ bo, float* __restrict__ out) {
  __shared__ __align__(16) bf16 as_[16 * kE];  // 12 KB

  const int tid = threadIdx.x;
  const int rt  = blockIdx.x;
  const bf16* Ab = A + (size_t)rt * 16 * kE;

#if HAVE_ASYNC_LDS
  for (int i = tid; i < (16 * kE) / 8; i += 256)   // 768 x b128 async copies
    async_cp16(Ab + i * 8, as_ + i * 8);
  wait_asynccnt<0>();
#else
  for (int i = tid; i < (16 * kE) / 8; i += 256)
    *(uint4*)(as_ + i * 8) = *(const uint4*)(Ab + i * 8);
#endif
  __syncthreads();

  const int w    = tid >> 5;
  const int lane = tid & 31;
  const int m    = lane & 15;
  const int half = lane >> 4;

  for (int ti = 0; ti < 3; ++ti) {             // 8 waves x 3 = 24 N tiles
    int   n0 = (w * 3 + ti) * 16;
    int   c  = n0 + m;
    float bv_ = bo[c];
    v8f acc = {bv_, bv_, bv_, bv_, bv_, bv_, bv_, bv_};

#pragma unroll
    for (int kk = 0; kk < kE; kk += 32) {
      v16bf a = ld2x8(as_ + m * kE + kk + half * 16,
                      as_ + m * kE + kk + half * 16 + 8);
      const bf16* wr = Wo + (size_t)c * kE + kk + half * 16;
      acc = wmma_bf16(a, ld2x8(wr, wr + 8), acc);
    }
#pragma unroll
    for (int r = 0; r < 8; ++r) {
      size_t R = (size_t)rt * 16 + r + half * 8;
      out[R * kE + c] = acc[r];
    }
  }
}

// ---- host ----------------------------------------------------------------

extern "C" void kernel_launch(void* const* d_in, const int* in_sizes, int n_in,
                              void* d_out, int out_size, void* d_ws, size_t ws_size,
                              hipStream_t stream) {
  const float*         X    = (const float*)d_in[0];
  const unsigned char* mask = (const unsigned char*)d_in[1];  // numpy bool
  const float* Wq = (const float*)d_in[2];
  const float* bq = (const float*)d_in[3];
  const float* Wk = (const float*)d_in[4];
  const float* bk = (const float*)d_in[5];
  const float* Wv = (const float*)d_in[6];
  const float* bv = (const float*)d_in[7];
  const float* Wo = (const float*)d_in[8];
  const float* bo = (const float*)d_in[9];

  char* ws = (char*)d_ws;
  const size_t wmat = (size_t)kE * kE;                 // 147456 elems
  bf16* Wqb = (bf16*)ws;
  bf16* Wkb = Wqb + wmat;
  bf16* Wvb = Wkb + wmat;
  bf16* Wob = Wvb + wmat;

  const size_t qkv_elems = (size_t)kB * kH * kS * kDP; // 4194304
  bf16* Qws  = Wob + wmat;
  bf16* Kws  = Qws + qkv_elems;
  bf16* Vtws = Kws + qkv_elems;
  bf16* Attn = Vtws + qkv_elems;                       // [B,S,E] bf16

  // zero Q/K/Vt (covers the D=48..63 padding lanes)
  (void)hipMemsetAsync(Qws, 0, 3 * qkv_elems * sizeof(bf16), stream);

  cvt_weights_kernel<<<(4 * (kE * kE / 4)) / 256, 256, 0, stream>>>(
      Wq, Wk, Wv, Wo, Wqb);

  qkv_kernel<<<(kB * kS) / 16, 256, 0, stream>>>(
      X, Wqb, Wkb, Wvb, bq, bk, bv, Qws, Kws, Vtws);

  attn_kernel<<<kB * kH * (kS / 128), 256, 0, stream>>>(
      Qws, Kws, Vtws, mask, Attn);

  oproj_kernel<<<(kB * kS) / 16, 256, 0, stream>>>(
      Attn, Wob, bo, (float*)d_out);
}
